// GatedDeltaNet_27814208209028
// MI455X (gfx1250) — matrix operand order repaired and checked by reading, not compiled, not run on hardware
//
#include <hip/hip_runtime.h>
#include <hip/hip_bf16.h>

// ---------------- problem constants ----------------
#define B_    2
#define S_    2048
#define BS_   (B_ * S_)       // 4096
#define HID_  1024
#define HK_   16
#define HV_   16
#define DK_   64
#define DV_   64
#define KCONV_ 4
#define KEY_DIM_ 1024
#define VAL_DIM_ 1024
#define CCH_  3072            // 2*KEY_DIM + VAL_DIM
#define ABCH_ 1056            // 2*HV + VAL_DIM

typedef __attribute__((ext_vector_type(16))) __bf16 v16bf;
typedef __attribute__((ext_vector_type(8)))  float  v8f;

__device__ __forceinline__ unsigned int bf16_rne(float f) {
  unsigned int u = __float_as_uint(f);
  return (u + 0x7FFFu + ((u >> 16) & 1u)) >> 16;   // round-to-nearest-even
}
__device__ __forceinline__ unsigned int pack_bf16x2(float lo, float hi) {
  return bf16_rne(lo) | (bf16_rne(hi) << 16);
}

union Frag {
  v16bf v;
  uint4 q[2];
};

// gfx1250 async global->LDS copy (16B per lane), tracked by ASYNCcnt.
__device__ __forceinline__ void async_ld16(void* lds_ptr, const void* gptr) {
  unsigned lds_off = (unsigned)(unsigned long long)lds_ptr;  // low 32b of flat LDS addr
  asm volatile("global_load_async_to_lds_b128 %0, %1, off"
               :: "v"(lds_off), "v"((unsigned long long)gptr) : "memory");
}
__device__ __forceinline__ void wait_async_le8() {
  asm volatile("s_wait_asynccnt 8" ::: "memory");
}
__device__ __forceinline__ void wait_async_le0() {
  asm volatile("s_wait_asynccnt 0" ::: "memory");
}

// =====================================================================
// f32 -> packed bf16 conversion (bandwidth-bound, 8 elems/thread)
// =====================================================================
__global__ __launch_bounds__(256)
void cvt_bf16_kernel(const float* __restrict__ src, unsigned short* __restrict__ dst,
                     long long n) {
  long long i0 = ((long long)blockIdx.x * 256 + threadIdx.x) * 8;
  if (i0 + 8 <= n) {
    float4 a = *(const float4*)(src + i0);
    float4 b = *(const float4*)(src + i0 + 4);
    uint4 o;
    o.x = pack_bf16x2(a.x, a.y);
    o.y = pack_bf16x2(a.z, a.w);
    o.z = pack_bf16x2(b.x, b.y);
    o.w = pack_bf16x2(b.z, b.w);
    *(uint4*)(dst + i0) = o;
  } else {
    for (long long i = i0; i < n; ++i)
      dst[i] = (unsigned short)bf16_rne(src[i]);
  }
}

// =====================================================================
// GEMM: C[M,N](f32) = A[M,K](bf16 rm) * W[N,K](bf16 rm)^T
// Block 256 threads (8 waves), block tile M=128 x N=128, K-step=64.
// Waves: 4 (M) x 2 (N); wave tile 32(M) x 64(N): 2 M-frags x 4 N-frags,
// each B-fragment feeds TWO consecutive WMMAs -> 16 WMMAs / wave / K-step.
// Double-buffered LDS staged with global_load_async_to_lds_b128, pipelined
// via ASYNCcnt (8 chunks/thread/tile; per-wave async loads complete in
// order => s_wait_asynccnt 8 guarantees the current tile is resident while
// the next tile's DMA is in flight).
// =====================================================================
#define TM 128
#define TN 128
#define TK 64

__global__ __launch_bounds__(256)
void gemm_bf16_wmma(const unsigned short* __restrict__ A,
                    const unsigned short* __restrict__ W,
                    float* __restrict__ C, int M, int N, int K) {
  __shared__ __align__(16) unsigned short As[2][TM][TK];   // 32 KB
  __shared__ __align__(16) unsigned short Bs[2][TN][TK];   // 32 KB

  const int tid  = threadIdx.x;
  const int lane = tid & 31;
  const int wave = tid >> 5;
  const int wm   = wave & 3;        // wave M index (0..3) -> 32 rows each
  const int wn   = wave >> 2;       // wave N index (0..1) -> 64 cols each
  const int m0   = blockIdx.x * TM;
  const int n0   = blockIdx.y * TN;

  const int g    = lane >> 4;       // half-wave group (0/1)
  const int hrow = lane & 15;

  v8f acc[2][4];
  #pragma unroll
  for (int mi = 0; mi < 2; ++mi)
    #pragma unroll
    for (int j = 0; j < 4; ++j)
      #pragma unroll
      for (int e = 0; e < 8; ++e) acc[mi][j][e] = 0.0f;

  // Stage one K-tile into LDS buffer `buf` (exactly 8 asyncs per thread).
  // OOB B rows are clamped to row 0: garbage lands only in output columns
  // n >= N, which are never stored.
  auto stage = [&](int buf, int k0) {
    #pragma unroll
    for (int ch = tid; ch < (TM * TK) / 8; ch += 256) {    // 4 chunks
      int r = ch >> 3;
      int c = (ch & 7) * 8;
      async_ld16(&As[buf][r][c], A + (size_t)(m0 + r) * K + k0 + c);
    }
    #pragma unroll
    for (int ch = tid; ch < (TN * TK) / 8; ch += 256) {    // 4 chunks
      int r = ch >> 3;
      int c = (ch & 7) * 8;
      int n = n0 + r;
      if (n >= N) n = 0;                                   // clamp, no branch
      async_ld16(&Bs[buf][r][c], W + (size_t)n * K + k0 + c);
    }
  };

  const int nk = K / TK;
  stage(0, 0);

  for (int k = 0; k < nk; ++k) {
    const int cur = k & 1;
    if (k + 1 < nk) {
      stage(1 - cur, (k + 1) * TK);   // prefetch next tile (async DMA)
      wait_async_le8();               // current tile resident
    } else {
      wait_async_le0();
    }
    __syncthreads();

    #pragma unroll
    for (int ks = 0; ks < 2; ++ks) {
      // Two A fragments (M and M+16), each 16(M)x32(K):
      // lane<16: K 0..7 & 16..23; lane>=16: K 8..15 & 24..31
      Frag a[2];
      #pragma unroll
      for (int mi = 0; mi < 2; ++mi) {
        const unsigned short* ap = &As[cur][wm * 32 + mi * 16 + hrow][ks * 32];
        a[mi].q[0] = *(const uint4*)(ap + 8 * g);
        a[mi].q[1] = *(const uint4*)(ap + 16 + 8 * g);
      }
      #pragma unroll
      for (int j = 0; j < 4; ++j) {
        // B fragment 32(K)x16(N): lane = N column; lanes<16 K=0..7 & 16..23,
        // lanes>=16 K=8..15 & 24..31. Reused by both M-fragment WMMAs.
        Frag bfr;
        const unsigned short* bp =
            &Bs[cur][wn * 64 + j * 16 + hrow][ks * 32 + 8 * g];
        bfr.q[0] = *(const uint4*)bp;
        bfr.q[1] = *(const uint4*)(bp + 16);
        acc[0][j] = __builtin_amdgcn_wmma_f32_16x16x32_bf16(
            false, a[0].v, false, bfr.v, (short)0, acc[0][j], false, false);
        acc[1][j] = __builtin_amdgcn_wmma_f32_16x16x32_bf16(
            false, a[1].v, false, bfr.v, (short)0, acc[1][j], false, false);
      }
    }
    __syncthreads();
  }

  // C/D layout: VGPR r -> M = r + 8*(lane>>4), N = lane&15
  #pragma unroll
  for (int mi = 0; mi < 2; ++mi) {
    #pragma unroll
    for (int j = 0; j < 4; ++j) {
      int n = n0 + wn * 64 + j * 16 + (lane & 15);
      if (n < N) {
        #pragma unroll
        for (int r = 0; r < 8; ++r) {
          int m = m0 + wm * 32 + mi * 16 + r + 8 * (lane >> 4);
          C[(size_t)m * N + n] = acc[mi][j][r];
        }
      }
    }
  }
}

// =====================================================================
// Causal depthwise conv (K=4) + SiLU + per-head L2 norm of q,k;
// beta = sigmoid(b), g = -exp(A_log)*softplus(a + dt_bias).
// One block (256 threads) per token.
// =====================================================================
__global__ __launch_bounds__(256)
void conv_act_kernel(const float* __restrict__ qkv, const float* __restrict__ ab,
                     const float* __restrict__ w_conv, const float* __restrict__ A_log,
                     const float* __restrict__ dt_bias,
                     float* __restrict__ qn, float* __restrict__ kn,
                     float* __restrict__ vout, float* __restrict__ g_ws,
                     float* __restrict__ beta_ws) {
  const int bs = blockIdx.x;
  const int b  = bs / S_;
  const int s  = bs % S_;
  __shared__ float buf[CCH_];
  __shared__ float rq[HK_], rk[HK_];

  for (int c = threadIdx.x; c < CCH_; c += 256) {
    float acc = 0.0f;
    #pragma unroll
    for (int i = 0; i < KCONV_; ++i) {
      int sp = s - (KCONV_ - 1) + i;
      if (sp >= 0)
        acc += qkv[((size_t)(b * S_ + sp)) * CCH_ + c] * w_conv[c * KCONV_ + i];
    }
    buf[c] = acc / (1.0f + __expf(-acc));   // SiLU
  }
  __syncthreads();

  if (threadIdx.x < 32) {
    int t = threadIdx.x;
    int h = t & 15;
    const float* base = &buf[(t < 16 ? 0 : KEY_DIM_) + h * DK_];
    float ss = 1e-6f;
    #pragma unroll 8
    for (int i = 0; i < DK_; ++i) ss += base[i] * base[i];
    float r = rsqrtf(ss);
    if (t < 16) rq[h] = r; else rk[h] = r;
  }
  if (threadIdx.x >= 32 && threadIdx.x < 48) {
    int h = threadIdx.x - 32;
    float av = ab[(size_t)bs * ABCH_ + h];
    float bv = ab[(size_t)bs * ABCH_ + HV_ + h];
    float beta = 1.0f / (1.0f + __expf(-bv));
    float x = av + dt_bias[h];
    float sp = (x > 20.0f) ? x : log1pf(__expf(x));
    g_ws[bs * HV_ + h]    = -__expf(A_log[h]) * sp;
    beta_ws[bs * HV_ + h] = beta;
  }
  __syncthreads();

  for (int c = threadIdx.x; c < KEY_DIM_; c += 256) {
    int h = c >> 6;
    qn[(size_t)bs * KEY_DIM_ + c]   = buf[c] * rq[h];
    kn[(size_t)bs * KEY_DIM_ + c]   = buf[KEY_DIM_ + c] * rk[h];
    vout[(size_t)bs * VAL_DIM_ + c] = buf[2 * KEY_DIM_ + c];
  }
}

// =====================================================================
// Gated delta-rule recurrence. One block per (batch, head); state
// h[64][64] in registers: thread (kg, v) owns h[kg*16..+16][v].
// =====================================================================
__global__ __launch_bounds__(256)
void scan_kernel(const float* __restrict__ qn, const float* __restrict__ kn,
                 const float* __restrict__ v, const float* __restrict__ g_ws,
                 const float* __restrict__ beta_ws, float* __restrict__ o_raw) {
  const int bh = blockIdx.x;
  const int b  = bh >> 4;
  const int h  = bh & 15;
  const int t  = threadIdx.x;
  const int vcol = t & 63;
  const int kg   = t >> 6;

  __shared__ float ks[DK_], qs[DK_], vs[DV_];
  __shared__ float red_kv[4][DV_], red_o[4][DV_];
  __shared__ float sg, sbeta;

  float hreg[16];
  #pragma unroll
  for (int i = 0; i < 16; ++i) hreg[i] = 0.0f;

  for (int s = 0; s < S_; ++s) {
    size_t base = ((size_t)(b * S_ + s) * HK_ + h) * DK_;
    if (t < 64)       ks[t]       = kn[base + t];
    else if (t < 128) qs[t - 64]  = qn[base + (t - 64)];
    else if (t < 192) vs[t - 128] = v[base + (t - 128)];
    else if (t == 192) sg    = g_ws[(b * S_ + s) * HV_ + h];
    else if (t == 193) sbeta = beta_ws[(b * S_ + s) * HV_ + h];
    __syncthreads();

    const float eg = __expf(sg);
    float partial = 0.0f;
    #pragma unroll
    for (int i = 0; i < 16; ++i) {
      hreg[i] *= eg;
      partial += ks[kg * 16 + i] * hreg[i];
    }
    red_kv[kg][vcol] = partial;
    __syncthreads();

    float kv = red_kv[0][vcol] + red_kv[1][vcol] + red_kv[2][vcol] + red_kv[3][vcol];
    float delta = sbeta * (vs[vcol] - kv);
    float op = 0.0f;
    #pragma unroll
    for (int i = 0; i < 16; ++i) {
      hreg[i] += ks[kg * 16 + i] * delta;
      op += qs[kg * 16 + i] * hreg[i];
    }
    red_o[kg][vcol] = op;
    __syncthreads();

    if (kg == 0) {
      o_raw[base + vcol] = 0.125f *
          (red_o[0][vcol] + red_o[1][vcol] + red_o[2][vcol] + red_o[3][vcol]);
    }
  }
}

// =====================================================================
// o = o * silu(gate); RMS-norm over DV per head; writes bf16 directly
// (feeds the output-projection WMMA GEMM).
// =====================================================================
__global__ __launch_bounds__(256)
void gate_norm_kernel(const float* __restrict__ o_raw, const float* __restrict__ ab,
                      const float* __restrict__ rms_w,
                      unsigned short* __restrict__ o_bf) {
  const int bs = blockIdx.x;
  __shared__ float buf[VAL_DIM_];
  __shared__ float rr[HV_];

  for (int c = threadIdx.x; c < VAL_DIM_; c += 256) {
    float gt = ab[(size_t)bs * ABCH_ + 2 * HV_ + c];
    float sgt = gt / (1.0f + __expf(-gt));
    buf[c] = o_raw[(size_t)bs * VAL_DIM_ + c] * sgt;
  }
  __syncthreads();
  if (threadIdx.x < HV_) {
    int h = threadIdx.x;
    float ss = 0.0f;
    #pragma unroll 8
    for (int i = 0; i < DV_; ++i) { float x = buf[h * DV_ + i]; ss += x * x; }
    rr[h] = rsqrtf(ss / (float)DV_ + 1e-5f);
  }
  __syncthreads();
  for (int c = threadIdx.x; c < VAL_DIM_; c += 256) {
    float y = buf[c] * rr[c >> 6] * rms_w[c & 63];
    o_bf[(size_t)bs * VAL_DIM_ + c] = (unsigned short)bf16_rne(y);
  }
}

// =====================================================================
extern "C" void kernel_launch(void* const* d_in, const int* in_sizes, int n_in,
                              void* d_out, int out_size, void* d_ws, size_t ws_size,
                              hipStream_t stream) {
  (void)in_sizes; (void)n_in; (void)out_size; (void)ws_size;
  const float* hs      = (const float*)d_in[0];
  const float* w_qkv   = (const float*)d_in[1];
  const float* w_ab    = (const float*)d_in[2];
  const float* w_conv  = (const float*)d_in[3];
  const float* A_log   = (const float*)d_in[4];
  const float* dt_bias = (const float*)d_in[5];
  const float* rms_w   = (const float*)d_in[6];
  const float* w_o     = (const float*)d_in[7];
  float* out = (float*)d_out;

  float* ws = (float*)d_ws;
  size_t off = 0;
  float* qkv = ws + off; off += (size_t)BS_ * CCH_;
  float* ab  = ws + off; off += (size_t)BS_ * ABCH_;
  float* qn  = ws + off; off += (size_t)BS_ * KEY_DIM_;
  float* kn  = ws + off; off += (size_t)BS_ * KEY_DIM_;
  float* vv  = ws + off; off += (size_t)BS_ * VAL_DIM_;
  float* gws = ws + off; off += (size_t)BS_ * HV_;
  float* bws = ws + off; off += (size_t)BS_ * HV_;
  // bf16 regions (ushort), carved from float workspace (counts are even)
  unsigned short* hs_bf   = (unsigned short*)(ws + off); off += (size_t)BS_ * HID_ / 2;
  unsigned short* wqkv_bf = (unsigned short*)(ws + off); off += (size_t)CCH_ * HID_ / 2;
  unsigned short* wab_bf  = (unsigned short*)(ws + off); off += (size_t)ABCH_ * HID_ / 2;
  unsigned short* wo_bf   = (unsigned short*)(ws + off); off += (size_t)HID_ * VAL_DIM_ / 2;
  unsigned short* on_bf   = (unsigned short*)(ws + off); off += (size_t)BS_ * VAL_DIM_ / 2;
  // qkv f32 buffer is dead after conv_act_kernel -> reuse for o_raw
  float* o_raw = qkv;

  dim3 blk(256);
  auto cvt_grid = [](long long n) { return dim3((unsigned)((n / 8 + 255) / 256)); };

  // f32 -> bf16 (weights once, activations once)
  cvt_bf16_kernel<<<cvt_grid((long long)BS_ * HID_), blk, 0, stream>>>(
      hs, hs_bf, (long long)BS_ * HID_);
  cvt_bf16_kernel<<<cvt_grid((long long)CCH_ * HID_), blk, 0, stream>>>(
      w_qkv, wqkv_bf, (long long)CCH_ * HID_);
  cvt_bf16_kernel<<<cvt_grid((long long)ABCH_ * HID_), blk, 0, stream>>>(
      w_ab, wab_bf, (long long)ABCH_ * HID_);
  cvt_bf16_kernel<<<cvt_grid((long long)HID_ * VAL_DIM_), blk, 0, stream>>>(
      w_o, wo_bf, (long long)HID_ * VAL_DIM_);

  // projections (WMMA)
  gemm_bf16_wmma<<<dim3(BS_ / TM, (CCH_ + TN - 1) / TN), blk, 0, stream>>>(
      hs_bf, wqkv_bf, qkv, BS_, CCH_, HID_);
  gemm_bf16_wmma<<<dim3(BS_ / TM, (ABCH_ + TN - 1) / TN), blk, 0, stream>>>(
      hs_bf, wab_bf, ab, BS_, ABCH_, HID_);

  conv_act_kernel<<<dim3(BS_), blk, 0, stream>>>(
      qkv, ab, w_conv, A_log, dt_bias, qn, kn, vv, gws, bws);

  scan_kernel<<<dim3(B_ * HV_), blk, 0, stream>>>(qn, kn, vv, gws, bws, o_raw);

  gate_norm_kernel<<<dim3(BS_), blk, 0, stream>>>(o_raw, ab, rms_w, on_bf);

  // output projection (WMMA)
  gemm_bf16_wmma<<<dim3(BS_ / TM, (HID_ + TN - 1) / TN), blk, 0, stream>>>(
      on_bf, wo_bf, out, BS_, HID_, VAL_DIM_);
}
